// SpatialNeuralProcess_20426864460200
// MI455X (gfx1250) — compile-verified, matrix operand order checked by reading, not run on hardware
//
#include <hip/hip_runtime.h>

// ---------------------------------------------------------------------------
// SpatialNeuralProcess on gfx1250 (MI455X, wave32, WMMA).
// All GEMM-shaped math routed through v_wmma_f32_16x16x32_f16.
// B operands are produced column-major (N x K) so both A and B tiles stage
// into LDS with b128 vector copies (no transpose scatter in the hot loop).
// ---------------------------------------------------------------------------

#define BS    16
#define NC    1024
#define NT    2048
#define XDIM  8
#define HDIM  256
#define HHALF 128
#define DDIM  384
#define MTOT  (BS * NT)

typedef __attribute__((ext_vector_type(16))) _Float16 v16h;
typedef __attribute__((ext_vector_type(8)))  float    v8f;

union Frag16 { v16h h; unsigned u[8]; };

// ---------------------------------------------------------------------------
// Generic gated WMMA GEMM:   Out = cast_f16( opt_relu( sum_e gate[row,e] *
//                            (A(f16) @ B_e(f16) + bias_e) ) )
// A: row-major M x K (ldA).  B: COLUMN-major, stored as BT[n][k] (ldB = K ld).
// Tile: 128(M) x 128(N) x 32(K); 256 threads = 8 waves, wave grid 4x2,
// each wave owns 2x4 16x16 f32 accumulator tiles.
// Fragment layouts follow CDNA5 ISA 7.12.2 (16-bit A 16x32, mirrored for B).
// Software pipeline: next K-slab global loads issued before current compute.
// ---------------------------------------------------------------------------
__global__ __launch_bounds__(256) void k_moe_gemm(
    const _Float16* __restrict__ A, long long sAb, int ldA,
    const _Float16* __restrict__ Bt, long long sBb, long long sBe, int ldB,
    const float* __restrict__ bias, int biasE,
    const float* __restrict__ gate,                 // [ (z*M+row)*E + e ], null => 1
    _Float16* __restrict__ Out, long long sOb, int ldOut,
    int M, int N, int K, int E, int relu)
{
  __shared__ __align__(16) _Float16 As[128 * 40];   // [m][k], stride 40
  __shared__ __align__(16) _Float16 Bs[128 * 40];   // [n][k], stride 40

  const int tid  = threadIdx.x;
  const int lane = tid & 31;
  const int wave = tid >> 5;
  const int wy   = wave & 3;          // 4 waves along M
  const int wx   = wave >> 2;         // 2 waves along N
  const int hlf  = lane >> 4;         // lane half selects K-group (ISA layout)
  const int l16  = lane & 15;
  const int bm0  = blockIdx.y * 128;
  const int bn0  = blockIdx.x * 128;
  const int z    = blockIdx.z;

  const _Float16* Ab = A + (size_t)z * sAb;
  _Float16*       Ob = Out + (size_t)z * sOb;

  const int trow = tid >> 1, tseg = tid & 1;        // 128 rows x 2 x 16 halves

  int kidx[8];
  #pragma unroll
  for (int j = 0; j < 8; ++j)
    kidx[j] = ((j & 4) << 2) + ((j & 3) << 1) + (hlf << 3);  // K index per frag VGPR

  float accf[2][4][8];
  #pragma unroll
  for (int a = 0; a < 2; ++a)
    #pragma unroll
    for (int b2 = 0; b2 < 4; ++b2)
      #pragma unroll
      for (int c = 0; c < 8; ++c) accf[a][b2][c] = 0.f;

  for (int e = 0; e < E; ++e) {
    const _Float16* Be = Bt + (size_t)z * sBb + (size_t)e * sBe;
    const _Float16* gA = Ab + (size_t)(bm0 + trow) * ldA + tseg * 16;
    const _Float16* gB = Be + (size_t)(bn0 + trow) * ldB + tseg * 16;

    v8f acc[2][4];
    v8f vz = {0.f, 0.f, 0.f, 0.f, 0.f, 0.f, 0.f, 0.f};
    #pragma unroll
    for (int a = 0; a < 2; ++a)
      #pragma unroll
      for (int b2 = 0; b2 < 4; ++b2) acc[a][b2] = vz;

    // register-staged pipeline: tile k held in ra/rb while computing tile k-1
    uint4 ra0 = ((const uint4*)gA)[0], ra1 = ((const uint4*)gA)[1];
    uint4 rb0 = ((const uint4*)gB)[0], rb1 = ((const uint4*)gB)[1];

    for (int k0 = 0; k0 < K; k0 += 32) {
      __syncthreads();                       // LDS free to overwrite
      {
        uint4* sa = (uint4*)&As[trow * 40 + tseg * 16];
        sa[0] = ra0; sa[1] = ra1;
        uint4* sb = (uint4*)&Bs[trow * 40 + tseg * 16];
        sb[0] = rb0; sb[1] = rb1;
      }
      __syncthreads();                       // tile published

      if (k0 + 32 < K) {                     // issue next slab early
        const uint4* na = (const uint4*)(gA + k0 + 32);
        ra0 = na[0]; ra1 = na[1];
        const uint4* nb = (const uint4*)(gB + k0 + 32);
        rb0 = nb[0]; rb1 = nb[1];
        if (k0 + 64 < K) {                   // keep L2 one more tile ahead
          __builtin_prefetch(gA + k0 + 64, 0, 1);
          __builtin_prefetch(gB + k0 + 64, 0, 1);
        }
      }

      Frag16 af[2], bf[4];
      #pragma unroll
      for (int mi = 0; mi < 2; ++mi) {
        int m = wy * 32 + mi * 16 + l16;
        #pragma unroll
        for (int j = 0; j < 8; ++j)
          af[mi].u[j] = *(const unsigned*)&As[m * 40 + kidx[j]];
      }
      #pragma unroll
      for (int ni = 0; ni < 4; ++ni) {
        int n = wx * 64 + ni * 16 + l16;
        #pragma unroll
        for (int j = 0; j < 8; ++j)
          bf[ni].u[j] = *(const unsigned*)&Bs[n * 40 + kidx[j]];
      }
      #pragma unroll
      for (int mi = 0; mi < 2; ++mi)
        #pragma unroll
        for (int ni = 0; ni < 4; ++ni)
          acc[mi][ni] = __builtin_amdgcn_wmma_f32_16x16x32_f16(
              false, af[mi].h, false, bf[ni].h, (short)0, acc[mi][ni], false, false);
    }

    // gated fold (C/D layout: n = lane&15, m = j + 8*(lane>>4))
    float bv[4];
    #pragma unroll
    for (int ni = 0; ni < 4; ++ni) {
      int col = bn0 + wx * 64 + ni * 16 + l16;
      bv[ni] = bias ? bias[e * biasE + col] : 0.f;
    }
    #pragma unroll
    for (int mi = 0; mi < 2; ++mi)
      #pragma unroll
      for (int j = 0; j < 8; ++j) {
        int row = bm0 + wy * 32 + mi * 16 + j + 8 * hlf;
        float g = gate ? gate[((size_t)z * M + row) * E + e] : 1.f;
        #pragma unroll
        for (int ni = 0; ni < 4; ++ni)
          accf[mi][ni][j] += g * (acc[mi][ni][j] + bv[ni]);
      }
  }

  #pragma unroll
  for (int mi = 0; mi < 2; ++mi)
    #pragma unroll
    for (int ni = 0; ni < 4; ++ni) {
      int col = bn0 + wx * 64 + ni * 16 + l16;
      #pragma unroll
      for (int j = 0; j < 8; ++j) {
        int row = bm0 + wy * 32 + mi * 16 + j + 8 * hlf;
        float v = accf[mi][ni][j];
        if (relu) v = v > 0.f ? v : 0.f;
        Ob[(size_t)row * ldOut + col] = (_Float16)v;
      }
    }
}

// ---------------------------------------------------------------------------
// fp32 -> fp16 weight conversion WITH transpose: dst[e][n][k] = src[e][k][n]
// ---------------------------------------------------------------------------
__global__ void k_f2h_t(const float* __restrict__ s, _Float16* __restrict__ d,
                        int Nn, int Kk, int total) {
  int i = blockIdx.x * 256 + threadIdx.x;
  if (i >= total) return;
  int e = i / (Nn * Kk);
  int r = i - e * (Nn * Kk);
  int n = r / Kk, k = r - n * Kk;
  d[i] = (_Float16)s[(size_t)e * Nn * Kk + (size_t)k * Nn + n];
}

// ---------------------------------------------------------------------------
// enc3[b][h]: mean_c(cxy @ W_proj3) + b_proj3, then @ W_lin + b_lin
// ---------------------------------------------------------------------------
__global__ __launch_bounds__(256) void k_prep(
    const float* __restrict__ cx, const float* __restrict__ cy,
    const float* __restrict__ W3, const float* __restrict__ b3,
    const float* __restrict__ Wl, const float* __restrict__ bl,
    float* __restrict__ enc3)
{
  __shared__ float tmp[HDIM];
  int b = blockIdx.x, h = threadIdx.x;
  float s = 0.f;
  for (int c = 0; c < NC; ++c) {
    const float* p = cx + ((size_t)b * NC + c) * XDIM;
    float v = cy[(size_t)b * NC + c] * W3[8 * HDIM + h];
    #pragma unroll
    for (int x = 0; x < XDIM; ++x) v += p[x] * W3[x * HDIM + h];
    s += v;
  }
  tmp[h] = s * (1.f / NC) + b3[h];
  __syncthreads();
  float o = bl[h];
  for (int i = 0; i < HDIM; ++i) o += tmp[i] * Wl[i * HDIM + h];
  enc3[b * HDIM + h] = o;
}

// ---------------------------------------------------------------------------
// h_e column-major for the GEMM B path: he[e][b][h][c] = cxy[b,c] @ W_exp_enc[e] + b
// ---------------------------------------------------------------------------
__global__ void k_embed(const float* __restrict__ cx, const float* __restrict__ cy,
                        const float* __restrict__ W, const float* __restrict__ bb,
                        _Float16* __restrict__ he)
{
  size_t tid = (size_t)blockIdx.x * 256 + threadIdx.x;   // [e][b][h][c]
  int c = tid & (NC - 1);
  int h = (tid >> 10) & 255;
  int b = (tid >> 18) & (BS - 1);
  int e = (int)(tid >> 22);
  const float* p = cx + ((size_t)b * NC + c) * XDIM;
  const float* w = W + (size_t)e * 9 * HDIM + h;
  float s = bb[e * HDIM + h] + cy[(size_t)b * NC + c] * w[8 * HDIM];
  #pragma unroll
  for (int x = 0; x < XDIM; ++x) s += p[x] * w[x * HDIM];
  he[tid] = (_Float16)s;
}

// ---------------------------------------------------------------------------
// simn[b][t][c] = exp(-||tx-cx||) / row_sum.  One wave per (b,t) row.
// ---------------------------------------------------------------------------
__global__ __launch_bounds__(256) void k_simn(const float* __restrict__ tx,
                                              const float* __restrict__ cx,
                                              _Float16* __restrict__ simn)
{
  int wave = threadIdx.x >> 5, lane = threadIdx.x & 31;
  int row = blockIdx.x * 8 + wave;          // b*NT + t
  int b = row >> 11;
  float p[XDIM];
  #pragma unroll
  for (int x = 0; x < XDIM; ++x) p[x] = tx[(size_t)row * XDIM + x];
  float sv[32];
  float sum = 0.f;
  #pragma unroll
  for (int i = 0; i < 32; ++i) {
    int c = lane + 32 * i;
    const float* q = cx + ((size_t)b * NC + c) * XDIM;
    float d2 = 0.f;
    #pragma unroll
    for (int x = 0; x < XDIM; ++x) { float d = p[x] - q[x]; d2 += d * d; }
    float s = __expf(-sqrtf(d2 + 1e-12f));
    sv[i] = s; sum += s;
  }
  #pragma unroll
  for (int o = 16; o > 0; o >>= 1) sum += __shfl_xor(sum, o, 32);
  float inv = 1.f / sum;
  #pragma unroll
  for (int i = 0; i < 32; ++i)
    simn[(size_t)row * NC + lane + 32 * i] = (_Float16)(sv[i] * inv);
}

// ---------------------------------------------------------------------------
// encoder gate softmax: target_x @ W_gate_enc + b, softmax over E=4
// ---------------------------------------------------------------------------
__global__ void k_gate_enc(const float* __restrict__ tx, const float* __restrict__ Wg,
                           const float* __restrict__ bg, float* __restrict__ gate)
{
  int row = blockIdx.x * 256 + threadIdx.x;
  const float* p = tx + (size_t)row * XDIM;
  float l[4];
  #pragma unroll
  for (int e = 0; e < 4; ++e) {
    float s = bg[e];
    #pragma unroll
    for (int x = 0; x < XDIM; ++x) s += p[x] * Wg[x * 4 + e];
    l[e] = s;
  }
  float mx = fmaxf(fmaxf(l[0], l[1]), fmaxf(l[2], l[3]));
  float den = 0.f;
  #pragma unroll
  for (int e = 0; e < 4; ++e) { l[e] = __expf(l[e] - mx); den += l[e]; }
  float inv = 1.f / den;
  #pragma unroll
  for (int e = 0; e < 4; ++e) gate[(size_t)row * 4 + e] = l[e] * inv;
}

// zcat cols [0,256): broadcast enc3 per batch (cols [256,512) filled by MoE GEMM)
__global__ void k_zcat(const float* __restrict__ enc3, _Float16* __restrict__ zcat) {
  size_t tid = (size_t)blockIdx.x * 256 + threadIdx.x;
  int h = tid & 255; size_t row = tid >> 8;
  zcat[row * 512 + h] = (_Float16)enc3[(row >> 11) * HDIM + h];
}

// zd cols [0,128): t_x = target_x @ W_attr + b_attr; cols [128,384): copy r
__global__ void k_zd(const float* __restrict__ tx, const float* __restrict__ Wa,
                     const float* __restrict__ ba, const _Float16* __restrict__ r,
                     _Float16* __restrict__ zd)
{
  size_t tid = (size_t)blockIdx.x * 256 + threadIdx.x;
  int col = (int)(tid % DDIM); size_t row = tid / DDIM;
  if (col < HHALF) {
    const float* p = tx + row * XDIM;
    float s = ba[col];
    #pragma unroll
    for (int x = 0; x < XDIM; ++x) s += p[x] * Wa[x * HHALF + col];
    zd[row * DDIM + col] = (_Float16)s;
  } else {
    zd[row * DDIM + col] = r[row * HDIM + (col - HHALF)];
  }
}

// decoder gate softmax over E=4 from zd (D=384)
__global__ void k_dgate(const _Float16* __restrict__ zd, const float* __restrict__ Wg,
                        const float* __restrict__ bg, float* __restrict__ gd)
{
  int row = blockIdx.x * 256 + threadIdx.x;
  float l[4] = {bg[0], bg[1], bg[2], bg[3]};
  for (int d = 0; d < DDIM; ++d) {
    float v = (float)zd[(size_t)row * DDIM + d];
    #pragma unroll
    for (int e = 0; e < 4; ++e) l[e] += v * Wg[d * 4 + e];
  }
  float mx = fmaxf(fmaxf(l[0], l[1]), fmaxf(l[2], l[3]));
  float den = 0.f;
  #pragma unroll
  for (int e = 0; e < 4; ++e) { l[e] = __expf(l[e] - mx); den += l[e]; }
  float inv = 1.f / den;
  #pragma unroll
  for (int e = 0; e < 4; ++e) gd[(size_t)row * 4 + e] = l[e] * inv;
}

// head: o = h2 @ W_d3 + b_d3; mu = o0; sigma = 0.1 + 0.9*softplus(o1)
__global__ void k_head(const _Float16* __restrict__ h2, const float* __restrict__ W3,
                       const float* __restrict__ b3, float* __restrict__ out,
                       float* __restrict__ muws)
{
  int row = blockIdx.x * 256 + threadIdx.x;
  float s0 = b3[0], s1 = b3[1];
  for (int i = 0; i < HDIM; ++i) {
    float v = (float)h2[(size_t)row * HDIM + i];
    s0 += v * W3[i * 2];
    s1 += v * W3[i * 2 + 1];
  }
  float sp = s1 > 20.f ? s1 : log1pf(__expf(s1));
  out[row] = s0;
  out[MTOT + row] = 0.1f + 0.9f * sp;
  muws[row] = s0;
}

// ---------------------------------------------------------------------------
// Moran k-NN spatial lag (k=5, decay 0.1).  One wave per (b,t) row; per-lane
// top-5 insertion over NT candidates, LDS merge, shared weights for y and mu.
// ---------------------------------------------------------------------------
__device__ __forceinline__ void ins5(float d, int idx, float* bd, int* bi) {
  if (d >= bd[4]) return;
  bd[4] = d; bi[4] = idx;
  #pragma unroll
  for (int q = 4; q > 0; --q)
    if (bd[q] < bd[q - 1]) {
      float td = bd[q]; bd[q] = bd[q - 1]; bd[q - 1] = td;
      int   ti = bi[q]; bi[q] = bi[q - 1]; bi[q - 1] = ti;
    }
}

__global__ __launch_bounds__(256) void k_moran(const float* __restrict__ tx,
                                               const float* __restrict__ ty,
                                               const float* __restrict__ mu,
                                               float* __restrict__ out)
{
  __shared__ float sd[8][160];
  __shared__ int   si[8][160];
  int wave = threadIdx.x >> 5, lane = threadIdx.x & 31;
  int row = blockIdx.x * 8 + wave;
  int b = row >> 11, t = row & (NT - 1);
  float p[XDIM];
  #pragma unroll
  for (int x = 0; x < XDIM; ++x) p[x] = tx[(size_t)row * XDIM + x];

  float bd[5] = {3e38f, 3e38f, 3e38f, 3e38f, 3e38f};
  int   bi[5] = {0, 0, 0, 0, 0};
  for (int c = lane; c < NT; c += 32) {
    const float* q = tx + ((size_t)b * NT + c) * XDIM;
    float d2 = 0.f;
    #pragma unroll
    for (int x = 0; x < XDIM; ++x) { float d = p[x] - q[x]; d2 += d * d; }
    float d = sqrtf(d2 + 1e-12f);
    if (c == t) d += 1e9f;                       // exclude self (eye * 1e9)
    ins5(d, c, bd, bi);
  }
  #pragma unroll
  for (int q = 0; q < 5; ++q) { sd[wave][lane * 5 + q] = bd[q]; si[wave][lane * 5 + q] = bi[q]; }
  __syncthreads();
  float fd[5] = {3e38f, 3e38f, 3e38f, 3e38f, 3e38f};
  int   fi[5] = {0, 0, 0, 0, 0};
  for (int i = 0; i < 160; ++i) ins5(sd[wave][i], si[wave][i], fd, fi);

  if (lane == 0) {
    float w[5], ws = 0.f;
    #pragma unroll
    for (int q = 0; q < 5; ++q) { w[q] = __expf(-0.1f * fd[q]); ws += w[q]; }
    float inv = 1.f / ws, ly = 0.f, lm = 0.f;
    #pragma unroll
    for (int q = 0; q < 5; ++q) {
      float wn = w[q] * inv;
      ly += wn * ty[(size_t)b * NT + fi[q]];
      lm += wn * mu[(size_t)b * NT + fi[q]];
    }
    out[2 * MTOT + row] = ly;
    out[3 * MTOT + row] = lm;
  }
}

// ---------------------------------------------------------------------------
extern "C" void kernel_launch(void* const* d_in, const int* in_sizes, int n_in,
                              void* d_out, int out_size, void* d_ws, size_t ws_size,
                              hipStream_t stream)
{
  (void)in_sizes; (void)n_in; (void)out_size; (void)ws_size;
  const float* context_x  = (const float*)d_in[0];
  const float* context_y  = (const float*)d_in[1];
  const float* target_x   = (const float*)d_in[2];
  const float* target_y   = (const float*)d_in[3];
  const float* W_gate_enc = (const float*)d_in[4];
  const float* b_gate_enc = (const float*)d_in[5];
  const float* W_exp_enc  = (const float*)d_in[6];
  const float* b_exp_enc  = (const float*)d_in[7];
  const float* W_proj3    = (const float*)d_in[8];
  const float* b_proj3    = (const float*)d_in[9];
  const float* W_lin      = (const float*)d_in[10];
  const float* b_lin      = (const float*)d_in[11];
  const float* W_fps1     = (const float*)d_in[12];
  const float* b_fps1     = (const float*)d_in[13];
  const float* W_fps2     = (const float*)d_in[14];
  const float* b_fps2     = (const float*)d_in[15];
  const float* W_attr     = (const float*)d_in[16];
  const float* b_attr     = (const float*)d_in[17];
  const float* W_gate_dec = (const float*)d_in[18];
  const float* b_gate_dec = (const float*)d_in[19];
  const float* W_exp_dec  = (const float*)d_in[20];
  const float* b_exp_dec  = (const float*)d_in[21];
  const float* W_d1       = (const float*)d_in[22];
  const float* b_d1       = (const float*)d_in[23];
  const float* W_d2       = (const float*)d_in[24];
  const float* b_d2       = (const float*)d_in[25];
  const float* W_d3       = (const float*)d_in[26];
  const float* b_d3       = (const float*)d_in[27];
  float* out = (float*)d_out;

  char* base = (char*)d_ws;
  size_t off = 0;
  auto carve = [&](size_t bytes) -> char* {
    char* p = base + off; off += (bytes + 255) & ~(size_t)255; return p;
  };

  _Float16* simn   = (_Float16*)carve((size_t)MTOT * NC * 2);        // 67 MB, reused below
  _Float16* he     = (_Float16*)carve((size_t)4 * BS * NC * HDIM * 2);
  _Float16* zcat   = (_Float16*)carve((size_t)MTOT * 512 * 2);
  _Float16* rbuf   = (_Float16*)carve((size_t)MTOT * HDIM * 2);
  _Float16* hid    = (_Float16*)carve((size_t)MTOT * HHALF * 2);
  float*    enc3   = (float*)carve((size_t)BS * HDIM * 4);
  float*    gateE  = (float*)carve((size_t)MTOT * 4 * 4);
  float*    gd     = (float*)carve((size_t)MTOT * 4 * 4);
  _Float16* wfps1t = (_Float16*)carve((size_t)128 * 512 * 2);     // [n][k]
  _Float16* wfps2t = (_Float16*)carve((size_t)256 * 128 * 2);
  _Float16* wexpdt = (_Float16*)carve((size_t)4 * DDIM * DDIM * 2);
  _Float16* wd1t   = (_Float16*)carve((size_t)256 * DDIM * 2);
  _Float16* wd2t   = (_Float16*)carve((size_t)256 * 256 * 2);
  // aliases into regions that are dead after the encoder MoE GEMM:
  _Float16* zd   = simn;
  _Float16* zd2  = simn + (size_t)MTOT * DDIM;
  float*    muws = (float*)(simn + (size_t)2 * MTOT * DDIM);
  _Float16* h1   = he;
  _Float16* h2   = he + (size_t)MTOT * HDIM;

  // weight precision conversion + transpose (tiny)
  k_f2h_t<<<256, 256, 0, stream>>>(W_fps1, wfps1t, 128, 512, 128 * 512);
  k_f2h_t<<<128, 256, 0, stream>>>(W_fps2, wfps2t, 256, 128, 256 * 128);
  k_f2h_t<<<2304, 256, 0, stream>>>(W_exp_dec, wexpdt, DDIM, DDIM, 4 * DDIM * DDIM);
  k_f2h_t<<<384, 256, 0, stream>>>(W_d1, wd1t, 256, DDIM, 256 * DDIM);
  k_f2h_t<<<256, 256, 0, stream>>>(W_d2, wd2t, 256, 256, 256 * 256);

  // encoder prep
  k_prep<<<BS, 256, 0, stream>>>(context_x, context_y, W_proj3, b_proj3, W_lin, b_lin, enc3);
  k_embed<<<65536, 256, 0, stream>>>(context_x, context_y, W_exp_enc, b_exp_enc, he);
  k_simn<<<MTOT / 8, 256, 0, stream>>>(target_x, context_x, simn);
  k_gate_enc<<<MTOT / 256, 256, 0, stream>>>(target_x, W_gate_enc, b_gate_enc, gateE);
  k_zcat<<<(MTOT * HDIM) / 256, 256, 0, stream>>>(enc3, zcat);

  // encoder MoE: per-b GEMM (2048x1024)x(1024x256) x 4 experts, gated fold
  k_moe_gemm<<<dim3(HDIM / 128, NT / 128, BS), 256, 0, stream>>>(
      simn, (long long)NT * NC, NC,
      he, (long long)HDIM * NC, (long long)BS * HDIM * NC, NC,
      nullptr, 0, gateE,
      zcat + HDIM, (long long)NT * 512, 512,
      NT, HDIM, NC, 4, 0);

  // fps MLP: relu(zcat @ W_fps1 + b) @ W_fps2 + b  -> r
  k_moe_gemm<<<dim3(1, MTOT / 128, 1), 256, 0, stream>>>(
      zcat, 0, 512, wfps1t, 0, 0, 512, b_fps1, 0, nullptr,
      hid, 0, HHALF, MTOT, HHALF, 512, 1, 1);
  k_moe_gemm<<<dim3(2, MTOT / 128, 1), 256, 0, stream>>>(
      hid, 0, HHALF, wfps2t, 0, 0, HHALF, b_fps2, 0, nullptr,
      rbuf, 0, HDIM, MTOT, HDIM, HHALF, 1, 0);

  // decoder input + gate
  k_zd<<<(MTOT * DDIM) / 256, 256, 0, stream>>>(target_x, W_attr, b_attr, rbuf, zd);
  k_dgate<<<MTOT / 256, 256, 0, stream>>>(zd, W_gate_dec, b_gate_dec, gd);

  // decoder MoE: (32768x384)x(384x384) x 4 experts, gated fold + expert bias
  k_moe_gemm<<<dim3(DDIM / 128, MTOT / 128, 1), 256, 0, stream>>>(
      zd, 0, DDIM, wexpdt, 0, (long long)DDIM * DDIM, DDIM, b_exp_dec, DDIM, gd,
      zd2, 0, DDIM, MTOT, DDIM, DDIM, 4, 0);

  // decoder MLP
  k_moe_gemm<<<dim3(2, MTOT / 128, 1), 256, 0, stream>>>(
      zd2, 0, DDIM, wd1t, 0, 0, DDIM, b_d1, 0, nullptr,
      h1, 0, HDIM, MTOT, HDIM, DDIM, 1, 1);
  k_moe_gemm<<<dim3(2, MTOT / 128, 1), 256, 0, stream>>>(
      h1, 0, HDIM, wd2t, 0, 0, HDIM, b_d2, 0, nullptr,
      h2, 0, HDIM, MTOT, HDIM, HDIM, 1, 1);

  // head + Moran lags
  k_head<<<MTOT / 256, 256, 0, stream>>>(h2, W_d3, b_d3, out, muws);
  k_moran<<<MTOT / 8, 256, 0, stream>>>(target_x, target_y, muws, out);
}
